// EnergyHead_72825465471144
// MI455X (gfx1250) — matrix operand order repaired and checked by reading, not compiled, not run on hardware
//
#include <hip/hip_runtime.h>

typedef __attribute__((ext_vector_type(16))) _Float16 v16h;
typedef __attribute__((ext_vector_type(8)))  _Float16 hv8;
typedef __attribute__((ext_vector_type(4)))  _Float16 hv4;
typedef __attribute__((ext_vector_type(8)))  float    v8f;
typedef __attribute__((ext_vector_type(4)))  float    f4v;

static constexpr int kGraphs  = 128;
static constexpr int kNodesPG = 512;
static constexpr int kC       = 512;   // channels (K)
static constexpr int kH       = 8;     // heads
static constexpr int kHID     = 256;   // hidden
static constexpr int kTileN   = 64;    // nodes per workgroup
static constexpr int kKB      = 64;    // K block staged in LDS
static constexpr int kXStr    = kKB + 8;  // 72 halfs = 144 B row stride (16B aligned, banks spread)
static constexpr int kWStr    = kKB + 8;  // 72 halfs

__device__ __forceinline__ float fast_rcp(float v) {
#if __has_builtin(__builtin_amdgcn_rcpf)
    return __builtin_amdgcn_rcpf(v);   // v_rcp_f32
#else
    return 1.0f / v;
#endif
}

// ---------------- prep: W1[h][k][n] f32  ->  W1T[h][n][k] f16 ----------------
__global__ __launch_bounds__(256)
void w1_transpose_kernel(const float* __restrict__ W1, _Float16* __restrict__ W1T)
{
    __shared__ float t[64 * 68];            // 64x64 f32 tile, pad 4 (272 B rows, 16B aligned)
    const int tid = threadIdx.x;
    const int bid = blockIdx.x;             // 8 heads * 8 kblk * 4 nblk = 256
    const int h   = bid >> 5;
    const int kb  = (bid >> 2) & 7;
    const int nb  = bid & 3;

    const float* src = W1 + ((size_t)h * kC + kb * 64) * kHID + nb * 64;
    #pragma unroll
    for (int i = 0; i < 4; ++i) {           // 4096 f32 / 256 thr = 4 float4
        int q = tid + 256 * i;
        int r = q >> 4, c4 = q & 15;
        f4v v = *(const f4v*)&src[(size_t)r * kHID + c4 * 4];
        *(f4v*)&t[r * 68 + c4 * 4] = v;
    }
    __syncthreads();
    #pragma unroll
    for (int i = 0; i < 2; ++i) {           // 4096 f16 out = 2 x hv8 per thread
        int q  = tid + 256 * i;
        int nl = q >> 3, g = q & 7;
        hv8 out;
        #pragma unroll
        for (int j = 0; j < 8; ++j)
            out[j] = (_Float16)t[(g * 8 + j) * 68 + nl];
        *(hv8*)&W1T[((size_t)h * kHID + nb * 64 + nl) * kC + kb * 64 + g * 8] = out;
    }
}

// ---------------- main: grouped MLP head with f16 WMMA ----------------
template <bool PRET>
__global__ __launch_bounds__(256)
void energy_head_f16(const float* __restrict__ x,
                     const int*   __restrict__ task,
                     const float* __restrict__ W1,
                     const _Float16* __restrict__ W1T,
                     const float* __restrict__ b1,
                     const float* __restrict__ W2,
                     const float* __restrict__ b2,
                     float*       __restrict__ y)
{
    __shared__ _Float16 xs[kTileN * kXStr];   //  9216 B
    __shared__ _Float16 w1s[kHID * kWStr];    // 36864 B (W1^T block: [n][k])
    __shared__ float    ysum[2][kTileN];      //   512 B

    const int tid  = threadIdx.x;
    const int wave = tid >> 5;      // wave32
    const int lane = tid & 31;
    const int lh   = lane >> 4;     // lane half
    const int ln   = lane & 15;

    const int tilesPerGraph = kNodesPG / kTileN;           // 8
    const int graph     = blockIdx.x / tilesPerGraph;
    const int jtile     = blockIdx.x % tilesPerGraph;
    const int node_base = graph * kNodesPG + jtile * kTileN;
    const int head      = task[graph];

    const int wm = wave & 3;        // 16-node row tile
    const int wn = wave >> 2;       // 128-col half

    v8f acc[8];
    {
        v8f z = {0.f, 0.f, 0.f, 0.f, 0.f, 0.f, 0.f, 0.f};
        #pragma unroll
        for (int t = 0; t < 8; ++t) acc[t] = z;
    }

    const float*    w1f = W1  + (size_t)head * kC * kHID;
    const _Float16* w1t = PRET ? (W1T + (size_t)head * kHID * kC) : nullptr;

    auto loadB = [](const _Float16* p) -> v16h {
        hv8 b0 = *(const hv8*)p;
        hv8 b1 = *(const hv8*)(p + 8);
        return __builtin_shufflevector(b0, b1, 0,1,2,3,4,5,6,7,8,9,10,11,12,13,14,15);
    };

    for (int kb = 0; kb < kC / kKB; ++kb) {
        const int kbase = kb * kKB;

        // ---- stage x tile [64 nodes x 64 k], f32 -> f16 ----
        #pragma unroll
        for (int i = 0; i < 4; ++i) {
            int q = tid + 256 * i;
            int r = q >> 4, c4 = q & 15;
            f4v v = *(const f4v*)&x[(size_t)(node_base + r) * kC + kbase + c4 * 4];
            hv4 h;
            h[0] = (_Float16)v[0]; h[1] = (_Float16)v[1];
            h[2] = (_Float16)v[2]; h[3] = (_Float16)v[3];
            *(hv4*)&xs[r * kXStr + c4 * 4] = h;
        }
        // ---- stage W1^T block [256 n x 64 k] ----
        if (PRET) {
            #pragma unroll
            for (int i = 0; i < 8; ++i) {      // 16384 f16 = 8 x hv8 per thread
                int q = tid + 256 * i;
                int n = q >> 3, g = q & 7;
                hv8 v = *(const hv8*)&w1t[(size_t)n * kC + kbase + g * 8];
                *(hv8*)&w1s[n * kWStr + g * 8] = v;
            }
        } else {
            // in-kernel transpose fallback: 4k x 4n micro-tiles
            #pragma unroll
            for (int i = 0; i < 4; ++i) {
                int q  = tid + 256 * i;
                int kg = q >> 6, n4 = q & 63;
                f4v r0 = *(const f4v*)&w1f[(size_t)(kbase + 4 * kg + 0) * kHID + n4 * 4];
                f4v r1 = *(const f4v*)&w1f[(size_t)(kbase + 4 * kg + 1) * kHID + n4 * 4];
                f4v r2 = *(const f4v*)&w1f[(size_t)(kbase + 4 * kg + 2) * kHID + n4 * 4];
                f4v r3 = *(const f4v*)&w1f[(size_t)(kbase + 4 * kg + 3) * kHID + n4 * 4];
                #pragma unroll
                for (int j = 0; j < 4; ++j) {
                    hv4 h;
                    h[0] = (_Float16)r0[j]; h[1] = (_Float16)r1[j];
                    h[2] = (_Float16)r2[j]; h[3] = (_Float16)r3[j];
                    *(hv4*)&w1s[(n4 * 4 + j) * kWStr + 4 * kg] = h;
                }
            }
        }
        __syncthreads();

        // prefetch next K block into WGP-level caches (global_prefetch_b8)
        if (kb + 1 < kC / kKB) {
            const int kn = kbase + kKB;
            __builtin_prefetch(&x[(size_t)(node_base + (tid >> 2)) * kC + kn + (tid & 3) * 16], 0, 3);
            if (PRET)
                __builtin_prefetch(&w1t[(size_t)(tid) * kC + kn], 0, 3);
            else
                __builtin_prefetch(&w1f[(size_t)(kn + (tid >> 3)) * kHID + (tid & 7) * 32], 0, 3);
        }

        // ---- compute: 2 k-steps of 32, 8 column tiles, B double-buffered ----
        #pragma unroll
        for (int ks = 0; ks < kKB / 32; ++ks) {
            const int k0 = ks * 32;
            const _Float16* xrow = &xs[(wm * 16 + ln) * kXStr];
            // A frag (contiguous k order): VGPR0-7 = k[k0+16*lh .. +16) (matches B)
            hv8 a0 = *(const hv8*)&xrow[k0 + 16 * lh];
            hv8 a1 = *(const hv8*)&xrow[k0 + 16 * lh + 8];
            v16h a = __builtin_shufflevector(a0, a1, 0,1,2,3,4,5,6,7,8,9,10,11,12,13,14,15);

            // B frag base for this wave: row n = wn*128 + t*16 + ln, k span [k0+16*lh, +16)
            const _Float16* wbase = &w1s[(wn * 128 + ln) * kWStr + k0 + 16 * lh];
            v16h bc = loadB(wbase);
            #pragma unroll
            for (int t = 0; t < 8; ++t) {
                v16h bn = bc;
                if (t < 7) bn = loadB(wbase + (size_t)(t + 1) * 16 * kWStr);  // prefetch next tile
                acc[t] = __builtin_amdgcn_wmma_f32_16x16x32_f16(
                    false, a, false, bc, (short)0, acc[t], false, false);
                bc = bn;
            }
        }
        __syncthreads();
    }

    // ---- epilogue: +b1, SiLU, dot with W2[head], column reduce (f32) ----
    float partial[8];
    #pragma unroll
    for (int v = 0; v < 8; ++v) partial[v] = 0.0f;

    #pragma unroll
    for (int t = 0; t < 8; ++t) {
        const int col = wn * 128 + t * 16 + ln;
        const float bv = b1[head * kHID + col];
        const float wv = W2[head * kHID + col];
        #pragma unroll
        for (int v = 0; v < 8; ++v) {
            float z = acc[t][v] + bv;
            float s = z * fast_rcp(1.0f + __expf(-z));   // silu: z*sigmoid(z)
            partial[v] += s * wv;
        }
    }
    #pragma unroll
    for (int v = 0; v < 8; ++v) {
        #pragma unroll
        for (int off = 1; off < 16; off <<= 1)
            partial[v] += __shfl_xor(partial[v], off, 32);
    }
    if (ln == 0) {   // lanes 0 and 16: nodes wm*16 + lh*8 + v  (disjoint slots, no atomics)
        #pragma unroll
        for (int v = 0; v < 8; ++v)
            ysum[wn][wm * 16 + lh * 8 + v] = partial[v];
    }
    __syncthreads();
    if (tid < kTileN)
        y[node_base + tid] = ysum[0][tid] + ysum[1][tid] + b2[head];
}

extern "C" void kernel_launch(void* const* d_in, const int* in_sizes, int n_in,
                              void* d_out, int out_size, void* d_ws, size_t ws_size,
                              hipStream_t stream) {
    (void)in_sizes; (void)n_in; (void)out_size;
    const float* x    = (const float*)d_in[0];
    // d_in[1] = n_node: uniform 512 per graph by construction
    const int*   task = (const int*)d_in[2];
    const float* W1   = (const float*)d_in[3];
    const float* b1   = (const float*)d_in[4];
    const float* W2   = (const float*)d_in[5];
    const float* b2   = (const float*)d_in[6];
    float* y = (float*)d_out;

    const dim3 grid(kGraphs * (kNodesPG / kTileN));   // 1024 workgroups
    const size_t need = (size_t)kH * kHID * kC * sizeof(_Float16);  // 2 MB

    if (d_ws && ws_size >= need) {
        _Float16* w1tg = (_Float16*)d_ws;
        w1_transpose_kernel<<<256, 256, 0, stream>>>(W1, w1tg);
        energy_head_f16<true><<<grid, 256, 0, stream>>>(x, task, W1, w1tg, b1, W2, b2, y);
    } else {
        energy_head_f16<false><<<grid, 256, 0, stream>>>(x, task, W1, nullptr, b1, W2, b2, y);
    }
}
